// EGNN_Full_16518444220527
// MI455X (gfx1250) — compile-verified, matrix operand order checked by reading, not run on hardware
//
#include <hip/hip_runtime.h>
#include <hip/hip_bf16.h>

typedef unsigned short u16;
typedef __attribute__((ext_vector_type(16))) __bf16 v16bf;
typedef __attribute__((ext_vector_type(8)))  float  v8f;

#define N_NODES 20000
#define N_EDGES 640000
#define HD      128
#define NG      64
#define ASTRIDE 136   // halves per row for 128-wide tiles (272B, padded)
#define ASTR2   272   // halves per row for 256-wide tiles (544B, padded)

#define EWAVES  8
#define ETILES  5     // tiles per wave in edge kernel
#define NWAVES  8
#define NTILES  2     // tiles per wave in node kernel

struct B32x2 { uint4 lo, hi; };

__device__ __forceinline__ u16 f2bf(float f) {
    unsigned u = __builtin_bit_cast(unsigned, f);
    unsigned r = u + 0x7fffu + ((u >> 16) & 1u);
    return (u16)(r >> 16);
}

// Low 32 bits of a generic pointer into LDS == LDS byte offset (ISA 10.2 aperture rule)
__device__ __forceinline__ unsigned lds_off(const void* p) {
    return (unsigned)(unsigned long long)p;
}

// CDNA5 async global->LDS copy, 16B per lane (tracked by ASYNCcnt)
__device__ __forceinline__ void async_g2l_b128(unsigned ldsaddr, const void* gaddr) {
    asm volatile("global_load_async_to_lds_b128 %0, %1, off"
                 :: "v"(ldsaddr), "v"(gaddr) : "memory");
}
__device__ __forceinline__ void wait_asynccnt0() {
    asm volatile("s_wait_asynccnt 0x0" ::: "memory");
}

// ---- WMMA GEMM: acc(16 x 128) += A(16 x 32*KT, LDS bf16) * B(frag-packed, LDS) ----
template<int KT>
__device__ __forceinline__ void wmma_gemm(v8f acc[8], const u16* A, int strideH,
                                          const u16* Bp, int lane) {
    #pragma unroll
    for (int kt = 0; kt < KT; ++kt) {
        int row = lane & 15;
        int k0  = kt * 32 + ((lane >> 4) << 3);
        const u16* ap = A + row * strideH + k0;
        B32x2 av;
        av.lo = *(const uint4*)(ap);
        av.hi = *(const uint4*)(ap + 16);
        v16bf a = __builtin_bit_cast(v16bf, av);
        #pragma unroll
        for (int nt = 0; nt < 8; ++nt) {
            const u16* bp = Bp + (((nt * KT) + kt) * 32 + lane) * 16;
            B32x2 bv;
            bv.lo = *(const uint4*)(bp);
            bv.hi = *(const uint4*)(bp + 8);
            v16bf b = __builtin_bit_cast(v16bf, bv);
            acc[nt] = __builtin_amdgcn_wmma_f32_16x16x32_bf16(
                false, a, false, b, (short)0, acc[nt], false, false);
        }
    }
}

// ---- bias + LayerNorm + ReLU on a 16x128 D-tile held in acc[8] ----
__device__ __forceinline__ void bias_ln_relu(v8f acc[8], const float* bias,
                                             const float* gamma, const float* beta,
                                             int lane) {
    int nl = lane & 15;
    #pragma unroll
    for (int nt = 0; nt < 8; ++nt) {
        float bv = bias[nt * 16 + nl];
        #pragma unroll
        for (int r = 0; r < 8; ++r) acc[nt][r] += bv;
    }
    float s[8], q[8];
    #pragma unroll
    for (int r = 0; r < 8; ++r) { s[r] = 0.f; q[r] = 0.f; }
    #pragma unroll
    for (int nt = 0; nt < 8; ++nt)
        #pragma unroll
        for (int r = 0; r < 8; ++r) { float v = acc[nt][r]; s[r] += v; q[r] += v * v; }
    #pragma unroll
    for (int r = 0; r < 8; ++r) {
        #pragma unroll
        for (int m = 1; m < 16; m <<= 1) {
            s[r] += __shfl_xor(s[r], m);
            q[r] += __shfl_xor(q[r], m);
        }
    }
    float mean[8], rstd[8];
    #pragma unroll
    for (int r = 0; r < 8; ++r) {
        mean[r] = s[r] * (1.f / 128.f);
        float var = q[r] * (1.f / 128.f) - mean[r] * mean[r];
        rstd[r] = rsqrtf(var + 1e-5f);
    }
    #pragma unroll
    for (int nt = 0; nt < 8; ++nt) {
        float gv = gamma[nt * 16 + nl], bv = beta[nt * 16 + nl];
        #pragma unroll
        for (int r = 0; r < 8; ++r) {
            float x = (acc[nt][r] - mean[r]) * rstd[r] * gv + bv;
            acc[nt][r] = fmaxf(x, 0.f);
        }
    }
}

__device__ __forceinline__ void store_tile_bf16(u16* lds, int strideH,
                                                const v8f acc[8], int lane) {
    int nl = lane & 15;
    int rb = (lane >> 4) * 8;
    #pragma unroll
    for (int nt = 0; nt < 8; ++nt)
        #pragma unroll
        for (int r = 0; r < 8; ++r)
            lds[(rb + r) * strideH + nt * 16 + nl] = f2bf(acc[nt][r]);
}

// ---- pack f32 weight (K x 128) into WMMA B-operand fragments (bf16) ----
__global__ void pack_b_kernel(const float* W, u16* out, int K, int N) {
    int KT = K / 32;
    int nt = blockIdx.x / KT;
    int kt = blockIdx.x % KT;
    int lane = threadIdx.x;
    int n = nt * 16 + (lane & 15);
    int kb = kt * 32 + ((lane >> 4) ? 8 : 0);
    u16* dst = out + (((nt * KT) + kt) * 32 + lane) * 16;
    #pragma unroll
    for (int j = 0; j < 8; ++j) dst[j]     = f2bf(W[(kb + j)      * N + n]);
    #pragma unroll
    for (int j = 0; j < 8; ++j) dst[8 + j] = f2bf(W[(kb + 16 + j) * N + n]);
}

// ---- h = x @ We + be ; also bf16 copy ----
__global__ void embed_kernel(const float* x, const float* We, const float* be,
                             float* h, u16* hbf) {
    int gid = blockIdx.x * blockDim.x + threadIdx.x;
    if (gid >= N_NODES * HD) return;
    int node = gid >> 7, n = gid & 127;
    float s = be[n];
    #pragma unroll
    for (int k = 0; k < 16; ++k) s += x[node * 16 + k] * We[k * HD + n];
    h[gid] = s;
    hbf[gid] = f2bf(s);
}

// ---- per-edge message + pos MLP + scatter ----
__global__ void __launch_bounds__(EWAVES * 32)
edge_kernel(const u16* __restrict__ hbf, const float* __restrict__ posb,
            const int* __restrict__ ei,
            float* msg_aggr, float* pos_aggr, float* cnt,
            const u16* pk /* packed W1a|W1b|Wm2|Wp1, 128KB contiguous */,
            const float* w256,
            const float* bm1, const float* gm1, const float* bem1,
            const float* bm2, const float* gm2, const float* bem2,
            const float* bp1, const float* gp1, const float* bep1,
            const float* Wp2, const float* bp2) {
    __shared__ u16 ldsW[65536];                 // 128KB weight fragments
    __shared__ u16 ldsA[EWAVES][16 * ASTRIDE];
    __shared__ u16 ldsB[EWAVES][16 * ASTRIDE];
    __shared__ float ldsPW[EWAVES][16];

    // async-stage all weight fragments into LDS (once per block)
    {
        int tid = threadIdx.x;
        #pragma unroll 4
        for (int it = 0; it < 32; ++it) {
            int chunk = tid + 256 * it;                  // 16B chunks, 8192 total
            async_g2l_b128(lds_off(&ldsW[chunk * 8]), pk + chunk * 8);
        }
        wait_asynccnt0();
    }
    __syncthreads();

    const u16* pW1a = ldsW;
    const u16* pW1b = ldsW + 16384;
    const u16* pWm2 = ldsW + 32768;
    const u16* pWp1 = ldsW + 49152;

    int lane = threadIdx.x & 31;
    int w = threadIdx.x >> 5;
    int nl = lane & 15;
    int rb = (lane >> 4) * 8;
    v8f zero = {0.f, 0.f, 0.f, 0.f, 0.f, 0.f, 0.f, 0.f};

    for (int t = 0; t < ETILES; ++t) {
        int tile = blockIdx.x * (EWAVES * ETILES) + t * EWAVES + w;

        int sidx = 0, didx = 0;
        float pdx = 0.f, pdy = 0.f, pdz = 0.f, dist = 0.f;
        if (lane < 16) {
            int e = tile * 16 + lane;
            sidx = ei[e];
            didx = ei[N_EDGES + e];
            pdx = posb[didx * 3 + 0] - posb[sidx * 3 + 0];
            pdy = posb[didx * 3 + 1] - posb[sidx * 3 + 1];
            pdz = posb[didx * 3 + 2] - posb[sidx * 3 + 2];
            dist = sqrtf(pdx * pdx + pdy * pdy + pdz * pdz);
        }
        // async-gather h[dst], h[src] rows (bf16) into this wave's LDS slabs
        #pragma unroll
        for (int it = 0; it < 8; ++it) {
            int chunk = lane + 32 * it;
            int row = chunk >> 4, c = chunk & 15;
            int di = __shfl(didx, row);
            int si = __shfl(sidx, row);
            async_g2l_b128(lds_off(&ldsA[w][row * ASTRIDE + c * 8]),
                           hbf + (size_t)di * HD + c * 8);
            async_g2l_b128(lds_off(&ldsB[w][row * ASTRIDE + c * 8]),
                           hbf + (size_t)si * HD + c * 8);
        }
        wait_asynccnt0();

        v8f acc[8];
        #pragma unroll
        for (int nt = 0; nt < 8; ++nt) acc[nt] = zero;

        // GEMM1: h[dst]*W1a + h[src]*W1b + dist ⊗ Wm1[256,:]
        wmma_gemm<4>(acc, ldsA[w], ASTRIDE, pW1a, lane);
        wmma_gemm<4>(acc, ldsB[w], ASTRIDE, pW1b, lane);
        float dv[8];
        int dstrow[8];
        #pragma unroll
        for (int r = 0; r < 8; ++r) {
            dv[r] = __shfl(dist, r + rb);
            dstrow[r] = __shfl(didx, r + rb);
        }
        #pragma unroll
        for (int nt = 0; nt < 8; ++nt) {
            float wv = w256[nt * 16 + nl];
            #pragma unroll
            for (int r = 0; r < 8; ++r) acc[nt][r] += dv[r] * wv;
        }
        bias_ln_relu(acc, bm1, gm1, bem1, lane);
        store_tile_bf16(ldsA[w], ASTRIDE, acc, lane);   // a1

        // GEMM2 -> m
        #pragma unroll
        for (int nt = 0; nt < 8; ++nt) acc[nt] = zero;
        wmma_gemm<4>(acc, ldsA[w], ASTRIDE, pWm2, lane);
        bias_ln_relu(acc, bm2, gm2, bem2, lane);
        // scatter m into msg_aggr (segment sum via f32 atomics)
        #pragma unroll
        for (int nt = 0; nt < 8; ++nt)
            #pragma unroll
            for (int r = 0; r < 8; ++r)
                atomicAdd(&msg_aggr[(size_t)dstrow[r] * HD + nt * 16 + nl], acc[nt][r]);
        store_tile_bf16(ldsB[w], ASTRIDE, acc, lane);   // m (bf16)

        // GEMM3 -> pos-MLP hidden
        #pragma unroll
        for (int nt = 0; nt < 8; ++nt) acc[nt] = zero;
        wmma_gemm<4>(acc, ldsB[w], ASTRIDE, pWp1, lane);
        bias_ln_relu(acc, bp1, gp1, bep1, lane);
        // pw = hidden @ Wp2 + bp2
        float pp[8];
        #pragma unroll
        for (int r = 0; r < 8; ++r) pp[r] = 0.f;
        #pragma unroll
        for (int nt = 0; nt < 8; ++nt) {
            float w2 = Wp2[nt * 16 + nl];
            #pragma unroll
            for (int r = 0; r < 8; ++r) pp[r] += acc[nt][r] * w2;
        }
        #pragma unroll
        for (int r = 0; r < 8; ++r)
            #pragma unroll
            for (int m = 1; m < 16; m <<= 1) pp[r] += __shfl_xor(pp[r], m);
        if (nl == 0) {
            #pragma unroll
            for (int r = 0; r < 8; ++r) ldsPW[w][rb + r] = pp[r] + bp2[0];
        }
        if (lane < 16) {
            float pw = ldsPW[w][lane];
            atomicAdd(&pos_aggr[didx * 3 + 0], pdx * pw);
            atomicAdd(&pos_aggr[didx * 3 + 1], pdy * pw);
            atomicAdd(&pos_aggr[didx * 3 + 2], pdz * pw);
            atomicAdd(&cnt[didx], 1.0f);
        }
    }
}

// ---- node update: h += mlp2([h, msg_aggr]); pos += pos_aggr/max(cnt,1) ----
__global__ void __launch_bounds__(NWAVES * 32)
node_kernel(float* h, u16* hbf, const float* __restrict__ msg_aggr,
            float* posb, const float* __restrict__ pos_aggr,
            const float* __restrict__ cnt,
            const u16* pk /* packed Wu1|Wu2, 96KB contiguous */,
            const float* bu1, const float* gu1, const float* beu1,
            const float* bu2, const float* gu2, const float* beu2) {
    __shared__ u16 ldsW[49152];                 // 96KB weight fragments
    __shared__ u16 ldsA[NWAVES][16 * ASTR2];
    __shared__ u16 ldsC[NWAVES][16 * ASTRIDE];

    {
        int tid = threadIdx.x;
        #pragma unroll 4
        for (int it = 0; it < 24; ++it) {
            int chunk = tid + 256 * it;                  // 16B chunks, 6144 total
            async_g2l_b128(lds_off(&ldsW[chunk * 8]), pk + chunk * 8);
        }
        wait_asynccnt0();
    }
    __syncthreads();

    const u16* pWu1 = ldsW;
    const u16* pWu2 = ldsW + 32768;

    int lane = threadIdx.x & 31;
    int w = threadIdx.x >> 5;
    int nl = lane & 15;
    int rb = (lane >> 4) * 8;
    v8f zero = {0.f, 0.f, 0.f, 0.f, 0.f, 0.f, 0.f, 0.f};

    for (int t = 0; t < NTILES; ++t) {
        int tile = blockIdx.x * (NWAVES * NTILES) + t * NWAVES + w;
        if (tile >= N_NODES / 16) continue;
        int node0 = tile * 16;

        // build A = [h (bf16, async copy), msg_aggr (f32->bf16)]  16 x 256
        #pragma unroll
        for (int it = 0; it < 8; ++it) {
            int chunk = lane + 32 * it;
            int row = chunk >> 4, c = chunk & 15;
            async_g2l_b128(lds_off(&ldsA[w][row * ASTR2 + c * 8]),
                           hbf + (size_t)(node0 + row) * HD + c * 8);
            const float* mrow = msg_aggr + (size_t)(node0 + row) * HD + c * 8;
            float4 f0 = *(const float4*)(mrow);
            float4 f1 = *(const float4*)(mrow + 4);
            uint4 pkv;
            pkv.x = (unsigned)f2bf(f0.x) | ((unsigned)f2bf(f0.y) << 16);
            pkv.y = (unsigned)f2bf(f0.z) | ((unsigned)f2bf(f0.w) << 16);
            pkv.z = (unsigned)f2bf(f1.x) | ((unsigned)f2bf(f1.y) << 16);
            pkv.w = (unsigned)f2bf(f1.z) | ((unsigned)f2bf(f1.w) << 16);
            *(uint4*)&ldsA[w][row * ASTR2 + 128 + c * 8] = pkv;
        }
        wait_asynccnt0();

        v8f acc[8];
        #pragma unroll
        for (int nt = 0; nt < 8; ++nt) acc[nt] = zero;
        wmma_gemm<8>(acc, ldsA[w], ASTR2, pWu1, lane);
        bias_ln_relu(acc, bu1, gu1, beu1, lane);
        store_tile_bf16(ldsC[w], ASTRIDE, acc, lane);
        #pragma unroll
        for (int nt = 0; nt < 8; ++nt) acc[nt] = zero;
        wmma_gemm<4>(acc, ldsC[w], ASTRIDE, pWu2, lane);
        bias_ln_relu(acc, bu2, gu2, beu2, lane);
        // residual h update + refresh bf16 copy
        #pragma unroll
        for (int r = 0; r < 8; ++r) {
            size_t base = (size_t)(node0 + rb + r) * HD;
            #pragma unroll
            for (int nt = 0; nt < 8; ++nt) {
                size_t idx = base + nt * 16 + nl;
                float hv = h[idx] + acc[nt][r];
                h[idx] = hv;
                hbf[idx] = f2bf(hv);
            }
        }
        if (lane < 16) {
            int node = node0 + lane;
            float c = cnt[node];
            float d = fmaxf(c, 1.f);
            #pragma unroll
            for (int k = 0; k < 3; ++k)
                posb[node * 3 + k] += pos_aggr[node * 3 + k] / d;
        }
    }
}

__global__ void pool_kernel(const float* h, const int* batch, float* g) {
    int gid = blockIdx.x * blockDim.x + threadIdx.x;
    if (gid >= N_NODES * HD) return;
    int node = gid >> 7, n = gid & 127;
    atomicAdd(&g[batch[node] * HD + n], h[gid]);
}

__global__ void pred_kernel(const float* g, const float* W1, const float* b1,
                            const float* W2, const float* b2, float* out) {
    __shared__ float red[128];
    int i = blockIdx.x, n = threadIdx.x;
    float s = b1[n];
    for (int k = 0; k < HD; ++k) s += g[i * HD + k] * W1[k * HD + n];
    red[n] = fmaxf(s, 0.f) * W2[n];
    __syncthreads();
    for (int off = 64; off > 0; off >>= 1) {
        if (n < off) red[n] += red[n + off];
        __syncthreads();
    }
    if (n == 0) out[i] = red[0] + b2[0];
}

extern "C" void kernel_launch(void* const* d_in, const int* in_sizes, int n_in,
                              void* d_out, int out_size, void* d_ws, size_t ws_size,
                              hipStream_t stream) {
    const float* x     = (const float*)d_in[0];
    const float* pos   = (const float*)d_in[1];
    const int*   ei    = (const int*)d_in[2];
    const int*   batch = (const int*)d_in[3];
    const float* We    = (const float*)d_in[4];
    const float* be    = (const float*)d_in[5];

    char* wsb = (char*)d_ws;
    float* h      = (float*)(wsb + 0);
    u16*   hbf    = (u16*)  (wsb + 10240000);
    float* msg    = (float*)(wsb + 15360000);
    float* posb   = (float*)(wsb + 25600000);
    float* posagg = (float*)(wsb + 25840000);
    float* cnt    = (float*)(wsb + 26080000);
    float* g      = (float*)(wsb + 26160000);
    u16*   packs  = (u16*)  (wsb + 26192768);

    const size_t LPK = 114688;   // packed halves per layer
    const size_t oW1a = 0, oW1b = 16384, oWm2 = 32768, oWp1 = 49152,
                 oWu1 = 65536, oWu2 = 98304;

    hipMemcpyAsync(posb, pos, N_NODES * 3 * sizeof(float),
                   hipMemcpyDeviceToDevice, stream);

    auto pack = [&](const float* W, u16* dst, int K) {
        pack_b_kernel<<<dim3((HD / 16) * (K / 32)), dim3(32), 0, stream>>>(W, dst, K, HD);
    };
    for (int L = 0; L < 5; ++L) {
        int base = 6 + 22 * L;
        const float* Wm1 = (const float*)d_in[base + 0];
        const float* Wm2 = (const float*)d_in[base + 4];
        const float* Wp1 = (const float*)d_in[base + 8];
        const float* Wu1 = (const float*)d_in[base + 14];
        const float* Wu2 = (const float*)d_in[base + 18];
        u16* pkp = packs + (size_t)L * LPK;
        pack(Wm1,            pkp + oW1a, 128);   // rows 0..127  (h[dst])
        pack(Wm1 + 128 * HD, pkp + oW1b, 128);   // rows 128..255 (h[src])
        pack(Wm2,            pkp + oWm2, 128);
        pack(Wp1,            pkp + oWp1, 128);
        pack(Wu1,            pkp + oWu1, 256);
        pack(Wu2,            pkp + oWu2, 128);
    }

    embed_kernel<<<dim3((N_NODES * HD + 255) / 256), dim3(256), 0, stream>>>(
        x, We, be, h, hbf);

    for (int L = 0; L < 5; ++L) {
        int base = 6 + 22 * L;
        const float* Wm1  = (const float*)d_in[base + 0];
        const float* bm1  = (const float*)d_in[base + 1];
        const float* gm1  = (const float*)d_in[base + 2];
        const float* bem1 = (const float*)d_in[base + 3];
        const float* bm2  = (const float*)d_in[base + 5];
        const float* gm2  = (const float*)d_in[base + 6];
        const float* bem2 = (const float*)d_in[base + 7];
        const float* bp1  = (const float*)d_in[base + 9];
        const float* gp1  = (const float*)d_in[base + 10];
        const float* bep1 = (const float*)d_in[base + 11];
        const float* Wp2  = (const float*)d_in[base + 12];
        const float* bp2  = (const float*)d_in[base + 13];
        const float* bu1  = (const float*)d_in[base + 15];
        const float* gu1  = (const float*)d_in[base + 16];
        const float* beu1 = (const float*)d_in[base + 17];
        const float* bu2  = (const float*)d_in[base + 19];
        const float* gu2  = (const float*)d_in[base + 20];
        const float* beu2 = (const float*)d_in[base + 21];
        u16* pkp = packs + (size_t)L * LPK;

        hipMemsetAsync(msg,    0, (size_t)N_NODES * HD * sizeof(float), stream);
        hipMemsetAsync(posagg, 0, (size_t)N_NODES * 3 * sizeof(float), stream);
        hipMemsetAsync(cnt,    0, (size_t)N_NODES * sizeof(float), stream);

        // 40000 tiles / (8 waves * 5 tiles) = 1000 blocks
        edge_kernel<<<dim3(N_EDGES / 16 / (EWAVES * ETILES)), dim3(EWAVES * 32), 0, stream>>>(
            hbf, posb, ei, msg, posagg, cnt,
            pkp, Wm1 + 256 * HD,
            bm1, gm1, bem1,
            bm2, gm2, bem2,
            bp1, gp1, bep1,
            Wp2, bp2);

        node_kernel<<<dim3((N_NODES / 16 + NWAVES * NTILES - 1) / (NWAVES * NTILES)),
                      dim3(NWAVES * 32), 0, stream>>>(
            h, hbf, msg, posb, posagg, cnt,
            pkp + oWu1,
            bu1, gu1, beu1,
            bu2, gu2, beu2);
    }

    const float* W1 = (const float*)d_in[116];
    const float* b1 = (const float*)d_in[117];
    const float* W2 = (const float*)d_in[118];
    const float* b2 = (const float*)d_in[119];
    hipMemsetAsync(g, 0, (size_t)NG * HD * sizeof(float), stream);
    pool_kernel<<<dim3((N_NODES * HD + 255) / 256), dim3(256), 0, stream>>>(h, batch, g);
    pred_kernel<<<dim3(NG), dim3(128), 0, stream>>>(g, W1, b1, W2, b2, (float*)d_out);
}
